// BaselineGRU_38070590112285
// MI455X (gfx1250) — matrix operand order repaired and checked by reading, not compiled, not run on hardware
//
#include <hip/hip_runtime.h>
#include <hip/hip_bf16.h>

typedef __attribute__((ext_vector_type(16))) _Float16 v16h;
typedef __attribute__((ext_vector_type(8)))  _Float16 h8;
typedef __attribute__((ext_vector_type(8)))  float    v8f;

// sizes from the reference
#define BB 8
#define NN 512
#define TT 256
#define DD 64
#define ROWS (BB * NN)          // 4096 logical GRU rows
#define GATES (3 * DD)          // 192

__device__ __forceinline__ float sigm(float x) {
    return __builtin_amdgcn_rcpf(1.f + __expf(-x));
}
__device__ __forceinline__ float tanh_fast(float x) {
    // tanh(x) = 2*sigmoid(2x) - 1
    return 2.f * __builtin_amdgcn_rcpf(1.f + __expf(-2.f * x)) - 1.f;
}

// ---- prep: u[o] = sum_k W_ih[o,k]*proj_w[k];  v[o] = sum_k W_ih[o,k]*proj_b[k] + b_ih[o]
__global__ void gru_prep_uv(const float* __restrict__ w_ih, const float* __restrict__ b_ih,
                            const float* __restrict__ proj_w, const float* __restrict__ proj_b,
                            float* __restrict__ uv) {
    int o = blockIdx.x * blockDim.x + threadIdx.x;
    if (o < GATES) {
        float u = 0.f, vv = 0.f;
        for (int k = 0; k < DD; ++k) {
            float w = w_ih[o * DD + k];
            u  += w * proj_w[k];
            vv += w * proj_b[k];
        }
        uv[o] = u;
        uv[GATES + o] = vv + b_ih[o];
    }
}

// ---- main: one wave32 per 16 rows; W_hh register-resident in B-layout; 24 WMMA/step
__global__ __launch_bounds__(32) void gru_main(
    const float* __restrict__ x,      // [B,N,T,1]
    const float* __restrict__ w_hh,   // [192,64]
    const float* __restrict__ b_hh,   // [192]
    const float* __restrict__ out_w,  // [1,64]
    const float* __restrict__ out_b,  // [1]
    const float* __restrict__ uv,     // [384] = {u[192], v[192]}
    float* __restrict__ out)          // [4096]
{
    __shared__ _Float16 hH[16 * DD];     // h staging, row-major [m][d], f16 (2 KB)
    __shared__ float    sX[TT * 16];     // per-step scalars, [j][m]       (16 KB)

    const int lane = threadIdx.x;
    const int lo   = lane & 15;
    const int hi   = lane >> 4;
    const int base = blockIdx.x * 16;    // first logical row of this wave

    // ---- preload the 16 rows' scalar sequences into LDS: sX[j*16 + m]
    for (int idx = lane; idx < TT * 16; idx += 32) {
        int m = idx & 15, j = idx >> 4;
        int i = base + m;
        int b = i >> 9;                  // i / 512
        int rem = i & 511;
        int t = rem >> 1;
        int nhi = rem & 1;
        // x flat index: ((b*512 + nhi*256 + j) * 256 + t)
        sX[idx] = x[(((b << 9) + (nhi << 8) + j) << 8) + t];
    }
    // h0 = 0
    for (int idx = lane; idx < 16 * DD; idx += 32) hH[idx] = (_Float16)0.f;

    // ---- per-lane gate constants: lane&15 indexes o within each 16-wide tile
    float uL[12], vL[12], bcL[4];
#pragma unroll
    for (int nt = 0; nt < 12; ++nt) {
        uL[nt] = uv[nt * 16 + lo];
        float vvv = uv[GATES + nt * 16 + lo];
        if (nt < 8) vvv += b_hh[nt * 16 + lo];   // fold b_hh into r/z constants
        vL[nt] = vvv;
    }
#pragma unroll
    for (int dt = 0; dt < 4; ++dt) bcL[dt] = b_hh[128 + dt * 16 + lo]; // c-gate b_hh (stays inside r*gh_c)

    // ---- load W_hh into registers in WMMA-B layout (constant across all 256 steps)
    // B[k,o] = W_hh[o,k]; lane holds col o = 16*nt + lo; element e <-> k = 32*kc + 16*hi + e
    v16h Bv[12][2];
#pragma unroll
    for (int nt = 0; nt < 12; ++nt) {
        int o = nt * 16 + lo;
#pragma unroll
        for (int kc = 0; kc < 2; ++kc) {
            const float4* p = (const float4*)&w_hh[o * DD + kc * 32 + hi * 16];
#pragma unroll
            for (int q = 0; q < 4; ++q) {
                float4 f = p[q];
                Bv[nt][kc][4 * q + 0] = (_Float16)f.x;
                Bv[nt][kc][4 * q + 1] = (_Float16)f.y;
                Bv[nt][kc][4 * q + 2] = (_Float16)f.z;
                Bv[nt][kc][4 * q + 3] = (_Float16)f.w;
            }
        }
    }

    // hidden state in f32 C-layout: H[dt] covers d in [16*dt,16*dt+16); element v <-> m = v + 8*hi
    v8f H[4];
#pragma unroll
    for (int dt = 0; dt < 4; ++dt) H[dt] = (v8f){};

    __syncthreads();   // single wave: cheap; orders LDS init vs loop

    for (int j = 0; j < TT; ++j) {
        // ---- rebuild A (h, f16, A-layout) from LDS
        // lane row m = lo; elements 0..15 <-> K = kbase + 16*hi + e
        v16h A[2];
#pragma unroll
        for (int kc = 0; kc < 2; ++kc) {
            h8 alo = *(const h8*)&hH[lo * DD + kc * 32 + hi * 16];
            h8 ahi = *(const h8*)&hH[lo * DD + kc * 32 + hi * 16 + 8];
            A[kc] = __builtin_shufflevector(alo, ahi, 0, 1, 2, 3, 4, 5, 6, 7,
                                                       8, 9, 10, 11, 12, 13, 14, 15);
        }
        // ---- broadcast this step's scalars: element v needs s[m = v + 8*hi]
        float sv[8];
#pragma unroll
        for (int v = 0; v < 8; ++v) sv[v] = sX[j * 16 + v + hi * 8];

        // ---- gh = h @ W_hh^T : 12 output tiles x (K=64 as 2 chunks) = 24 WMMAs
        v8f C[12];
#pragma unroll
        for (int nt = 0; nt < 12; ++nt) {
            v8f z = (v8f){};
            z = __builtin_amdgcn_wmma_f32_16x16x32_f16(false, A[0], false, Bv[nt][0],
                                                       (short)0, z, false, false);
            C[nt] = __builtin_amdgcn_wmma_f32_16x16x32_f16(false, A[1], false, Bv[nt][1],
                                                           (short)0, z, false, false);
        }

        // ---- gates + state update; write new h (f16) back to LDS for next step's A
#pragma unroll
        for (int dt = 0; dt < 4; ++dt) {
#pragma unroll
            for (int v = 0; v < 8; ++v) {
                float s  = sv[v];
                float r  = sigm(C[dt][v]     + s * uL[dt]     + vL[dt]);
                float zz = sigm(C[dt + 4][v] + s * uL[dt + 4] + vL[dt + 4]);
                float g  = (s * uL[dt + 8] + vL[dt + 8]) + r * (C[dt + 8][v] + bcL[dt]);
                float c  = tanh_fast(g);
                float hn = (1.f - zz) * c + zz * H[dt][v];
                H[dt][v] = hn;
                hH[(v + hi * 8) * DD + dt * 16 + lo] = (_Float16)hn;
            }
        }
        __syncthreads();   // single-wave barrier -> just orders DS store vs next load
    }

    // ---- output: out[i] = dot(h[i,:], out_w) + out_b
    float owL[4];
#pragma unroll
    for (int dt = 0; dt < 4; ++dt) owL[dt] = out_w[dt * 16 + lo];
    float ob = out_b[0];
#pragma unroll
    for (int v = 0; v < 8; ++v) {
        float p = 0.f;
#pragma unroll
        for (int dt = 0; dt < 4; ++dt) p += H[dt][v] * owL[dt];
        p += __shfl_xor(p, 1, 16);
        p += __shfl_xor(p, 2, 16);
        p += __shfl_xor(p, 4, 16);
        p += __shfl_xor(p, 8, 16);
        if (lo == 0) out[base + v + hi * 8] = p + ob;   // i -> (b= i/512, n = i%512) flat == i
    }
}

extern "C" void kernel_launch(void* const* d_in, const int* in_sizes, int n_in,
                              void* d_out, int out_size, void* d_ws, size_t ws_size,
                              hipStream_t stream) {
    const float* x      = (const float*)d_in[0];
    const float* proj_w = (const float*)d_in[1];
    const float* proj_b = (const float*)d_in[2];
    const float* w_ih   = (const float*)d_in[3];
    const float* w_hh   = (const float*)d_in[4];
    const float* b_ih   = (const float*)d_in[5];
    const float* b_hh   = (const float*)d_in[6];
    const float* out_w  = (const float*)d_in[7];
    const float* out_b  = (const float*)d_in[8];
    float* uv = (float*)d_ws;                    // 384 floats of scratch

    gru_prep_uv<<<1, 192, 0, stream>>>(w_ih, b_ih, proj_w, proj_b, uv);
    gru_main<<<ROWS / 16, 32, 0, stream>>>(x, w_hh, b_hh, out_w, out_b, uv, (float*)d_out);
}
